// MultiHeadAttention_29463475651228
// MI455X (gfx1250) — matrix operand order repaired and checked by reading, not compiled, not run on hardware
//
#include <hip/hip_runtime.h>

typedef __bf16 bf16;
typedef __attribute__((ext_vector_type(16))) __bf16 v16bf;
typedef __attribute__((ext_vector_type(8)))  float  v8f;
typedef __attribute__((vector_size(16)))     int    v4i;

#define DEV __device__ __forceinline__

constexpr int DMODEL = 1024;
constexpr int NHEADS = 16;
constexpr int DHEAD  = 64;
constexpr int SEQ    = 4096;
constexpr int BATCH  = 2;
constexpr int TOK    = BATCH * SEQ;            // 8192 rows for the big GEMMs
constexpr int MF     = 266;                    // real Performer features
constexpr int MFP    = 288;                    // padded to multiple of 32 for WMMA K
constexpr int CHUNK  = 128;
constexpr int NCH    = SEQ / CHUNK;            // 32 chunks
constexpr int RWS    = BATCH * NHEADS * SEQ;   // 131072 feature rows
constexpr float KEPS = 1e-4f;
constexpr float CEPS = 1e-6f;

#if __has_builtin(__builtin_amdgcn_global_load_async_to_lds_b128)
#define HAVE_ASYNC_LDS 1
#else
#define HAVE_ASYNC_LDS 0
#endif

// ---------------------------------------------------------------------------
// WMMA helpers (CDNA5 wave32, V_WMMA_F32_16X16X32_BF16)
// A fragment: lanes 0-15 hold M=lane, lanes 16-31 hold M=lane-16 with K+8;
//             elements 0..7 -> K base, 8..15 -> K base+16 (ISA 7.12.2).
// B fragment: stored as Bt[N][K] row-major; lane -> N, K split same as A fragment.
// C fragment: VGPR v -> M=v (+8 for hi lanes), N=lane&15.
// ---------------------------------------------------------------------------
DEV v8f wmma_bf16(v16bf a, v16bf b, v8f c) {
  return __builtin_amdgcn_wmma_f32_16x16x32_bf16(false, a, false, b, (short)0, c,
                                                 false, false);
}

DEV v16bf load_a_bf16(const bf16* A, int row0, int k0, int ld) {
  int l = threadIdx.x & 31;
  int m = l & 15;
  int kb = (l >> 4) * 8;
  const bf16* p = A + (long)(row0 + m) * ld + k0 + kb;
  v16bf a;
#pragma unroll
  for (int e = 0; e < 8; ++e) a[e] = p[e];
#pragma unroll
  for (int e = 0; e < 8; ++e) a[8 + e] = p[16 + e];
  return a;
}

// A fragment where the logical A[m][k] is stored transposed: Ac[k*ld + m]
DEV v16bf load_a_bf16_t(const bf16* Ac, int m0, int k0, int ld) {
  int l = threadIdx.x & 31;
  int m = (l & 15) + m0;
  int kb = (l >> 4) * 8 + k0;
  v16bf a;
#pragma unroll
  for (int e = 0; e < 8; ++e) a[e] = Ac[(long)(kb + e) * ld + m];
#pragma unroll
  for (int e = 0; e < 8; ++e) a[8 + e] = Ac[(long)(kb + 16 + e) * ld + m];
  return a;
}

DEV v16bf load_a_f32_scaled(const float* A, int row0, int k0, int ld, float s) {
  int l = threadIdx.x & 31;
  int m = l & 15;
  int kb = (l >> 4) * 8;
  const float* p = A + (long)(row0 + m) * ld + k0 + kb;
  v16bf a;
#pragma unroll
  for (int e = 0; e < 8; ++e) a[e] = (__bf16)(p[e] * s);
#pragma unroll
  for (int e = 0; e < 8; ++e) a[8 + e] = (__bf16)(p[16 + e] * s);
  return a;
}

// A rows are tokens t=b*SEQ+pos, columns k of the [b,h,pos,64] tensor (head gather)
DEV v16bf load_a_headgather(const bf16* A, int row0, int k0) {
  int l = threadIdx.x & 31;
  int m = l & 15;
  int kb = (l >> 4) * 8;
  int t = row0 + m;
  int bb = t / SEQ, pos = t - bb * SEQ;
  v16bf a;
#pragma unroll
  for (int half = 0; half < 2; ++half) {
    int ks = k0 + kb + 16 * half;           // multiple of 8 -> stays inside a head
    int h = ks >> 6, d = ks & 63;
    const bf16* p = A + ((long)((bb * NHEADS + h) * SEQ + pos) << 6) + d;
#pragma unroll
    for (int e = 0; e < 8; ++e) a[half * 8 + e] = p[e];
  }
  return a;
}

DEV v16bf load_b_bf16(const bf16* Bt, int n0, int k0, int ld) {
  int l = threadIdx.x & 31;
  int n = l & 15;
  int kb = (l >> 4) * 8;
  const bf16* p = Bt + (long)(n0 + n) * ld + k0 + kb;
  v16bf b;
#pragma unroll
  for (int e = 0; e < 8; ++e) b[e] = p[e];
#pragma unroll
  for (int e = 0; e < 8; ++e) b[8 + e] = p[16 + e];
  return b;
}

// ---------------------------------------------------------------------------
// Elementwise conversion kernels
// ---------------------------------------------------------------------------
__global__ void cvt_f32_to_bf16(const float* __restrict__ src, bf16* __restrict__ dst,
                                long n) {
  long i = (long)blockIdx.x * blockDim.x + threadIdx.x;
  if (i < n) dst[i] = (__bf16)src[i];
}

__global__ void cvt_proj_pad(const float* __restrict__ src, bf16* __restrict__ dst) {
  int i = blockIdx.x * blockDim.x + threadIdx.x;   // MFP*64
  if (i >= MFP * DHEAD) return;
  int m = i >> 6, d = i & 63;
  dst[i] = (__bf16)((m < MF) ? src[m * DHEAD + d] : 0.f);
}

__global__ void init_gmax(unsigned* g) { *g = 0x007FFFFFu; }  // enc(-inf)

// ---------------------------------------------------------------------------
// Generic GEMM: C[M,N] = A[M,K] @ Bt[N,K]^T + bias
// AMODE 0: row-major bf16; AMODE 1: head-gather from [b,h,pos,64]
// CMODE 0: row-major f32;  CMODE 1: scatter to [b,h,pos,64]
// 8 waves/block; each wave owns a 16x(16*NT) C strip: A fragment reused NT
// times per k-step, software-pipelined global loads.
// ---------------------------------------------------------------------------
template <int AMODE, int CMODE, int NT>
__global__ void gemm_wmma_bf16(const bf16* __restrict__ A, const bf16* __restrict__ Bt,
                               const float* __restrict__ bias, float* __restrict__ C,
                               int M, int N, int K) {
  int wave = threadIdx.y;
  int tn = blockIdx.x * (16 * NT);
  int tm = (blockIdx.y * 8 + wave) * 16;
  if (tm >= M) return;
  int l = threadIdx.x & 31;

  v8f acc[NT] = {};
  v16bf a = (AMODE == 0) ? load_a_bf16(A, tm, 0, K) : load_a_headgather(A, tm, 0);
  v16bf b[NT];
#pragma unroll
  for (int t = 0; t < NT; ++t) b[t] = load_b_bf16(Bt, tn + 16 * t, 0, K);

  for (int k0 = 0; k0 < K; k0 += 32) {
    v16bf an = a;
    v16bf bn[NT];
    if (k0 + 32 < K) {  // software pipeline: fetch next fragments first
      an = (AMODE == 0) ? load_a_bf16(A, tm, k0 + 32, K)
                        : load_a_headgather(A, tm, k0 + 32);
#pragma unroll
      for (int t = 0; t < NT; ++t) bn[t] = load_b_bf16(Bt, tn + 16 * t, k0 + 32, K);
      if (AMODE == 0)
        __builtin_prefetch(A + (long)(tm + (l & 15)) * K + k0 + 64, 0, 3);
    } else {
#pragma unroll
      for (int t = 0; t < NT; ++t) bn[t] = b[t];
    }
#pragma unroll
    for (int t = 0; t < NT; ++t) acc[t] = wmma_bf16(a, b[t], acc[t]);
    a = an;
#pragma unroll
    for (int t = 0; t < NT; ++t) b[t] = bn[t];
  }

  int mb = tm + ((l >> 4) << 3);
#pragma unroll
  for (int t = 0; t < NT; ++t) {
    int n = tn + 16 * t + (l & 15);
    float bv = bias ? bias[n] : 0.f;
#pragma unroll
    for (int v = 0; v < 8; ++v) {
      int m = mb + v;
      float val = acc[t][v] + bv;
      if (CMODE == 0) {
        C[(long)m * N + n] = val;
      } else {
        int bb = m / SEQ, pos = m - bb * SEQ;
        int h = n >> 6, d = n & 63;
        C[((long)((bb * NHEADS + h) * SEQ + pos) << 6) + d] = val;
      }
    }
  }
}

// ---------------------------------------------------------------------------
// Performer feature map: dash = (x * 64^-0.25) @ proj^T  (K = 64, NT=2 tiles)
// ---------------------------------------------------------------------------
__global__ void feat_dash_wmma(const float* __restrict__ Xh, const bf16* __restrict__ projT,
                               float* __restrict__ dash) {
  int wave = threadIdx.y;
  int tn = blockIdx.x * 32;                  // feature strip (2 tiles)
  int tm = (blockIdx.y * 8 + wave) * 16;     // row tile
  const float s = 0.35355339059327373f;      // 64^-0.25
  v8f acc[2] = {};
  for (int k0 = 0; k0 < DHEAD; k0 += 32) {
    v16bf a = load_a_f32_scaled(Xh, tm, k0, DHEAD, s);
#pragma unroll
    for (int t = 0; t < 2; ++t) {
      v16bf b = load_b_bf16(projT, tn + 16 * t, k0, DHEAD);
      acc[t] = wmma_bf16(a, b, acc[t]);
    }
  }
  int l = threadIdx.x & 31;
  int mb = tm + ((l >> 4) << 3);
#pragma unroll
  for (int t = 0; t < 2; ++t) {
    int n = tn + 16 * t + (l & 15);
#pragma unroll
    for (int v = 0; v < 8; ++v) dash[(long)(mb + v) * MFP + n] = acc[t][v];
  }
}

__global__ void feat_stats(const float* __restrict__ Xh, const float* __restrict__ dash,
                           float* __restrict__ diag, float* __restrict__ stab,
                           unsigned* __restrict__ gmax, int is_query) {
  long r = (long)blockIdx.x * blockDim.x + threadIdx.x;
  if (r >= RWS) return;
  const float s2 = 0.125f;                    // (64^-0.25)^2
  const float* q = Xh + r * DHEAD;
  float ss = 0.f;
  for (int d = 0; d < DHEAD; ++d) { float v = q[d]; ss += v * v; }
  diag[r] = 0.5f * s2 * ss;
  const float* dr = dash + r * MFP;
  float mx = -3.4e38f;
  for (int m = 0; m < MF; ++m) mx = fmaxf(mx, dr[m]);
  if (is_query) {
    stab[r] = mx;
  } else {
    unsigned u = __float_as_uint(mx);
    unsigned enc = (u & 0x80000000u) ? ~u : (u | 0x80000000u);  // order-preserving
    atomicMax(gmax, enc);
  }
}

__global__ void feat_exp(const float* __restrict__ dash, const float* __restrict__ diag,
                         const float* __restrict__ stab, const unsigned* __restrict__ gmax,
                         bf16* __restrict__ outp, int is_query) {
  long i = (long)blockIdx.x * blockDim.x + threadIdx.x;
  if (i >= (long)RWS * MFP) return;
  long r = i / MFP;
  int m = (int)(i - r * MFP);
  float val = 0.f;
  if (m < MF) {
    float st;
    if (is_query) {
      st = stab[r];
    } else {
      unsigned enc = *gmax;
      unsigned u = (enc & 0x80000000u) ? (enc ^ 0x80000000u) : ~enc;
      st = __uint_as_float(u);
    }
    const float ratio = 0.06131393394849658f;  // 266^-0.5
    val = ratio * (__expf(dash[i] - diag[r] - st) + KEPS);
  }
  outp[i] = (__bf16)val;
}

// ---------------------------------------------------------------------------
// Chunked causal linear attention. One workgroup per (b,h); 32 sequential
// chunks of 128 rows. Uses ~226KB of WGP LDS (CDNA5: up to 320KB/WG).
//   out = (Qp @ ctx_prev + tril(Qp Kp^T) @ V) / D
//   ctx += Kp^T @ V ; kstate += colsum(Kp)
// Chunk staging uses GLOBAL_LOAD_ASYNC_TO_LDS (ASYNCcnt) when available.
// ---------------------------------------------------------------------------
__global__ void causal_attn(const bf16* __restrict__ qp, const bf16* __restrict__ kp,
                            const float* __restrict__ Vf, bf16* __restrict__ attnb,
                            float* __restrict__ ctxf_ws, bf16* __restrict__ ctxb_ws) {
  extern __shared__ char smem[];
  bf16*  sQ     = (bf16*)(smem);            // [128][288] bf16   73728 B
  bf16*  sK     = (bf16*)(smem + 73728);    // [128][288] bf16   73728 B
  bf16*  sVT    = (bf16*)(smem + 147456);   // [64][128]  bf16   16384 B
  bf16*  sS     = (bf16*)(smem + 163840);   // [128][128] bf16   32768 B
  float* sO     = (float*)(smem + 196608);  // [128][64]  f32    32768 B
  float* kstate = (float*)(smem + 229376);  // [288]              1152 B
  float* dinv   = (float*)(smem + 230528);  // [128]               512 B
  float* rowq   = (float*)(smem + 231040);  // [128]               512 B

  int bh   = blockIdx.x;
  int tid  = threadIdx.y * 32 + threadIdx.x;
  int wave = threadIdx.y;
  int l    = threadIdx.x & 31;

  float* ctxf = ctxf_ws + (long)bh * DHEAD * MFP;   // ctx^T [d][m] f32
  bf16*  ctxb = ctxb_ws + (long)bh * DHEAD * MFP;   // ctx^T [d][m] bf16

  for (int i = tid; i < DHEAD * MFP; i += 256) { ctxf[i] = 0.f; ctxb[i] = (__bf16)0.f; }
  for (int i = tid; i < MFP; i += 256) kstate[i] = 0.f;
  __syncthreads();

  for (int c = 0; c < NCH; ++c) {
    long r0 = (long)bh * SEQ + c * CHUNK;
    const bf16* qsrc = qp + r0 * MFP;
    const bf16* ksrc = kp + r0 * MFP;
#if HAVE_ASYNC_LDS
    {
      v4i* gq = (v4i*)const_cast<bf16*>(qsrc);
      v4i* gk = (v4i*)const_cast<bf16*>(ksrc);
      v4i* lq = (v4i*)sQ;
      v4i* lk = (v4i*)sK;
      const int NB = CHUNK * MFP * 2 / 16;   // 4608 x 16B packets
      for (int i = tid; i < NB; i += 256) {
        __builtin_amdgcn_global_load_async_to_lds_b128(gq + i, lq + i, 0, 0);
        __builtin_amdgcn_global_load_async_to_lds_b128(gk + i, lk + i, 0, 0);
      }
      asm volatile("s_wait_asynccnt 0" ::: "memory");
    }
#else
    for (int i = tid; i < CHUNK * MFP; i += 256) { sQ[i] = qsrc[i]; sK[i] = ksrc[i]; }
#endif
    const float* vsrc = Vf + (r0 << 6);
    for (int i = tid; i < CHUNK * DHEAD; i += 256) {
      int j = i >> 6, d = i & 63;
      sVT[d * CHUNK + j] = (__bf16)vsrc[i];
    }
    __syncthreads();

    // --- phase 1: out_inter = Qp @ ctx_prev -> sO ; q.kstate / rowsum(q) ---
    {
      int m0 = wave * 16;
      v8f acc[4] = {};
      for (int k0 = 0; k0 < MFP; k0 += 32) {
        v16bf a = load_a_bf16(sQ, m0, k0, MFP);
#pragma unroll
        for (int ct = 0; ct < 4; ++ct) {
          v16bf b = load_b_bf16(ctxb, ct * 16, k0, MFP);
          acc[ct] = wmma_bf16(a, b, acc[ct]);
        }
      }
      int mb = m0 + ((l >> 4) << 3);
#pragma unroll
      for (int ct = 0; ct < 4; ++ct) {
        int n = ct * 16 + (l & 15);
#pragma unroll
        for (int v = 0; v < 8; ++v) sO[(mb + v) * DHEAD + n] = acc[ct][v];
      }
    }
    if (tid < CHUNK) {
      float di = 0.f, rq = 0.f;
      const bf16* qrow = sQ + tid * MFP;
      for (int m = 0; m < MFP; ++m) {
        float qv = (float)qrow[m];
        di += qv * kstate[m];
        rq += qv;
      }
      dinv[tid] = di;     // temporarily q . kstate
      rowq[tid] = rq;
    }
    __syncthreads();

    // --- phase 2: S = tril(Qp @ Kp^T) -> sS (bf16) ---
    {
      int m0 = wave * 16;
      v8f acc[8] = {};
      for (int k0 = 0; k0 < MFP; k0 += 32) {
        v16bf a = load_a_bf16(sQ, m0, k0, MFP);
#pragma unroll
        for (int ct = 0; ct < 8; ++ct) {
          if (ct <= wave) {               // uniform per wave: EXEC stays full
            v16bf b = load_b_bf16(sK, ct * 16, k0, MFP);
            acc[ct] = wmma_bf16(a, b, acc[ct]);
          }
        }
      }
      int mb = m0 + ((l >> 4) << 3);
#pragma unroll
      for (int ct = 0; ct < 8; ++ct) {
        int n = ct * 16 + (l & 15);
#pragma unroll
        for (int v = 0; v < 8; ++v) {
          int m = mb + v;
          sS[m * CHUNK + n] = (__bf16)((n <= m) ? acc[ct][v] : 0.f);
        }
      }
    }
    __syncthreads();

    // --- denominator: D = q.kstate + rowsum(S) + eps * rowsum(q) ---
    if (tid < CHUNK) {
      float s = 0.f;
      const bf16* srow = sS + tid * CHUNK;
      for (int j = 0; j < CHUNK; ++j) s += (float)srow[j];
      float D = dinv[tid] + s + CEPS * rowq[tid];
      dinv[tid] = 1.0f / D;
    }
    __syncthreads();

    // --- phase 3: out = (out_inter + S @ V) * Dinv -> attnb ---
    {
      int m0 = wave * 16;
      v8f acc[4] = {};
      for (int k0 = 0; k0 < CHUNK; k0 += 32) {
        v16bf a = load_a_bf16(sS, m0, k0, CHUNK);
#pragma unroll
        for (int ct = 0; ct < 4; ++ct) {
          v16bf b = load_b_bf16(sVT, ct * 16, k0, CHUNK);
          acc[ct] = wmma_bf16(a, b, acc[ct]);
        }
      }
      int mb = m0 + ((l >> 4) << 3);
#pragma unroll
      for (int ct = 0; ct < 4; ++ct) {
        int n = ct * 16 + (l & 15);
#pragma unroll
        for (int v = 0; v < 8; ++v) {
          int m = mb + v;
          float val = (sO[m * DHEAD + n] + acc[ct][v]) * dinv[m];
          attnb[((r0 + m) << 6) + n] = (__bf16)val;
        }
      }
    }

    // --- phase 4: ctx^T += (Kp^T @ V) ; kstate += colsum(Kp) ---
    {
      for (int tf = wave; tf < MFP / 16; tf += 8) {   // 18 feature tiles
        int tmf = tf * 16;
        v8f acc[4] = {};
        for (int k0 = 0; k0 < CHUNK; k0 += 32) {
          v16bf a = load_a_bf16_t(sK, tmf, k0, MFP);  // A[m][j] = Kp[j][m]
#pragma unroll
          for (int td = 0; td < 4; ++td) {
            v16bf b = load_b_bf16(sVT, td * 16, k0, CHUNK);
            acc[td] = wmma_bf16(a, b, acc[td]);
          }
        }
        int mb = tmf + ((l >> 4) << 3);
#pragma unroll
        for (int td = 0; td < 4; ++td) {
          int d = td * 16 + (l & 15);
#pragma unroll
          for (int v = 0; v < 8; ++v) {
            long idx = (long)d * MFP + (mb + v);
            float nv = ctxf[idx] + acc[td][v];
            ctxf[idx] = nv;
            ctxb[idx] = (__bf16)nv;
          }
        }
      }
      for (int m = tid; m < MFP; m += 256) {
        float s = 0.f;
        for (int j = 0; j < CHUNK; ++j) s += (float)sK[j * MFP + m];
        kstate[m] += s;
      }
    }
    __syncthreads();
  }
}

// ---------------------------------------------------------------------------
extern "C" void kernel_launch(void* const* d_in, const int* in_sizes, int n_in,
                              void* d_out, int out_size, void* d_ws, size_t ws_size,
                              hipStream_t stream) {
  (void)in_sizes; (void)n_in; (void)out_size; (void)ws_size;
  const float* x    = (const float*)d_in[0];
  const float* Wq   = (const float*)d_in[1];
  const float* bq   = (const float*)d_in[2];
  const float* Wk   = (const float*)d_in[3];
  const float* bk   = (const float*)d_in[4];
  const float* Wv   = (const float*)d_in[5];
  const float* bv   = (const float*)d_in[6];
  const float* Wo   = (const float*)d_in[7];
  const float* bo   = (const float*)d_in[8];
  const float* proj = (const float*)d_in[9];
  float* out = (float*)d_out;

  char* ws = (char*)d_ws;
  size_t off = 0;
  auto alloc = [&](size_t bytes) {
    size_t o = off;
    off = (off + bytes + 255) & ~(size_t)255;
    return o;
  };

  bf16*  XB    = (bf16*)(ws + alloc((size_t)TOK * DMODEL * 2));
  bf16*  WQB   = (bf16*)(ws + alloc((size_t)DMODEL * DMODEL * 2));
  bf16*  WKB   = (bf16*)(ws + alloc((size_t)DMODEL * DMODEL * 2));
  bf16*  WVB   = (bf16*)(ws + alloc((size_t)DMODEL * DMODEL * 2));
  bf16*  WOB   = (bf16*)(ws + alloc((size_t)DMODEL * DMODEL * 2));
  bf16*  PROJB = (bf16*)(ws + alloc((size_t)MFP * DHEAD * 2));
  float* Qf    = (float*)(ws + alloc((size_t)TOK * DMODEL * 4));
  float* Kf    = (float*)(ws + alloc((size_t)TOK * DMODEL * 4));
  float* Vfp   = (float*)(ws + alloc((size_t)TOK * DMODEL * 4));
  float* DASH  = (float*)(ws + alloc((size_t)RWS * MFP * 4));
  float* DIAG  = (float*)(ws + alloc((size_t)RWS * 4));
  float* STAB  = (float*)(ws + alloc((size_t)RWS * 4));
  unsigned* GMAX = (unsigned*)(ws + alloc(256));
  bf16*  QP    = (bf16*)(ws + alloc((size_t)RWS * MFP * 2));
  bf16*  KP    = (bf16*)(ws + alloc((size_t)RWS * MFP * 2));
  float* CTXF  = (float*)(ws + alloc((size_t)BATCH * NHEADS * DHEAD * MFP * 4));
  bf16*  CTXB  = (bf16*)(ws + alloc((size_t)BATCH * NHEADS * DHEAD * MFP * 2));
  bf16*  ATTNB = (bf16*)(ws + alloc((size_t)TOK * DMODEL * 2));

  // 1) bf16 conversions
  {
    long n = (long)TOK * DMODEL;
    cvt_f32_to_bf16<<<(unsigned)((n + 255) / 256), 256, 0, stream>>>(x, XB, n);
    long nw = (long)DMODEL * DMODEL;
    unsigned gw = (unsigned)((nw + 255) / 256);
    cvt_f32_to_bf16<<<gw, 256, 0, stream>>>(Wq, WQB, nw);
    cvt_f32_to_bf16<<<gw, 256, 0, stream>>>(Wk, WKB, nw);
    cvt_f32_to_bf16<<<gw, 256, 0, stream>>>(Wv, WVB, nw);
    cvt_f32_to_bf16<<<gw, 256, 0, stream>>>(Wo, WOB, nw);
    cvt_proj_pad<<<(MFP * DHEAD + 255) / 256, 256, 0, stream>>>(proj, PROJB);
  }

  dim3 blk(32, 8);  // 8 waves of 32 (wave32)

  // 2) Q/K/V projections, scattered to [b,h,pos,64]
  {
    dim3 grid(DMODEL / 64, TOK / 128);
    gemm_wmma_bf16<0, 1, 4><<<grid, blk, 0, stream>>>(XB, WQB, bq, Qf, TOK, DMODEL, DMODEL);
    gemm_wmma_bf16<0, 1, 4><<<grid, blk, 0, stream>>>(XB, WKB, bk, Kf, TOK, DMODEL, DMODEL);
    gemm_wmma_bf16<0, 1, 4><<<grid, blk, 0, stream>>>(XB, WVB, bv, Vfp, TOK, DMODEL, DMODEL);
  }

  // 3) Performer feature maps (reuse DASH buffer for q then k)
  {
    dim3 fgrid(MFP / 32, RWS / 128);
    long ne = (long)RWS * MFP;
    unsigned ge = (unsigned)((ne + 255) / 256);

    feat_dash_wmma<<<fgrid, blk, 0, stream>>>(Qf, PROJB, DASH);
    feat_stats<<<RWS / 256, 256, 0, stream>>>(Qf, DASH, DIAG, STAB, nullptr, 1);
    feat_exp<<<ge, 256, 0, stream>>>(DASH, DIAG, STAB, nullptr, QP, 1);

    init_gmax<<<1, 1, 0, stream>>>(GMAX);
    feat_dash_wmma<<<fgrid, blk, 0, stream>>>(Kf, PROJB, DASH);
    feat_stats<<<RWS / 256, 256, 0, stream>>>(Kf, DASH, DIAG, nullptr, GMAX, 0);
    feat_exp<<<ge, 256, 0, stream>>>(DASH, DIAG, nullptr, GMAX, KP, 0);
  }

  // 4) causal linear attention: one workgroup per (b,h), big-LDS path
  {
    size_t smem = 231552;  // fits CDNA5 320KB WGP LDS
    causal_attn<<<BATCH * NHEADS, blk, smem, stream>>>(QP, KP, Vfp, ATTNB, CTXF, CTXB);
  }

  // 5) output projection: gather heads back to [b,n,1024] and apply Wo^T + bo
  {
    dim3 grid(DMODEL / 64, TOK / 128);
    gemm_wmma_bf16<1, 0, 4><<<grid, blk, 0, stream>>>(ATTNB, WOB, bo, out, TOK, DMODEL, DMODEL);
  }
}
